// CustomBERTEncoderBlock_84215718740060
// MI455X (gfx1250) — matrix operand order repaired and checked by reading, not compiled, not run on hardware
//
#include <hip/hip_runtime.h>

// ---------------------------------------------------------------------------
// BERT encoder block for MI455X (gfx1250, wave32).
// All GEMMs run on v_wmma_f32_16x16x32_bf16 (fp32 accumulate), tiled
// 64x128x32 per 256-thread block (8 waves: 2 in M x 4 in N, 32x32 per wave).
// Operand B is pre-transposed to [N][K] so both A and B WMMA fragments are
// contiguous-K 128-bit LDS reads matching the CDNA5 VGPR layouts.
// ---------------------------------------------------------------------------

typedef __bf16 bf16;
typedef __attribute__((ext_vector_type(16))) __bf16 v16bf;
typedef __attribute__((ext_vector_type(8)))  __bf16 v8bf;
typedef __attribute__((ext_vector_type(8)))  float  v8f;
typedef __attribute__((ext_vector_type(4)))  float  v4f;

#define BM 64
#define BN 128
#define BK 32

// ------------------------- elementwise fp32 -> bf16 -------------------------
__global__ __launch_bounds__(256) void f32_to_bf16_kernel(
    const float* __restrict__ in, bf16* __restrict__ out, long long n) {
  long long i = ((long long)blockIdx.x * 256 + threadIdx.x) * 4;
  if (i + 3 < n) {
    v4f v = *(const v4f*)(in + i);
    out[i + 0] = (bf16)v[0];
    out[i + 1] = (bf16)v[1];
    out[i + 2] = (bf16)v[2];
    out[i + 3] = (bf16)v[3];
  }
}

// --------------------- tiled transpose + convert to bf16 --------------------
// in:  fp32 [R][C]  (+ z*inBatch)
// out: bf16 [C][R]  (+ z*outBatch)
__global__ __launch_bounds__(256) void transpose_convert_kernel(
    const float* __restrict__ in, bf16* __restrict__ out,
    int R, int C, long long inBatch, long long outBatch) {
  __shared__ float tile[32][33];
  const float* ip = in + (long long)blockIdx.z * inBatch;
  bf16* op = out + (long long)blockIdx.z * outBatch;
  const int tx = threadIdx.x & 31;
  const int ty = threadIdx.x >> 5;  // 0..7
  const int r0 = blockIdx.y * 32;
  const int c0 = blockIdx.x * 32;
#pragma unroll
  for (int i = 0; i < 4; ++i)
    tile[ty + 8 * i][tx] = ip[(long long)(r0 + ty + 8 * i) * C + (c0 + tx)];
  __syncthreads();
#pragma unroll
  for (int i = 0; i < 4; ++i)
    op[(long long)(c0 + ty + 8 * i) * R + (r0 + tx)] = (bf16)tile[tx][ty + 8 * i];
}

// ------------------------------- WMMA GEMM ----------------------------------
// C = scale * (A @ B) + bias, optional ReLU.
// A : bf16 [M][K] row-major        (+ z*aBatch)
// Bt: bf16 [N][K] row-major (B^T)  (+ z*bBatch)
// outF fp32 [M][N], outB bf16 [M][N] (either may be null) (+ z*cBatch)
__device__ __forceinline__ v16bf frag_a(const bf16* base, int lh) {
  // A 16x32 bf16 layout: lanes 0-15 M=0..15; VGPR0-3 = K 8h..8h+7,
  // VGPR4-7 = K 16+8h..16+8h+7  (h = lane>>4)
  v8bf lo = *(const v8bf*)(base + 8 * lh);
  v8bf hi = *(const v8bf*)(base + 16 + 8 * lh);
  return __builtin_shufflevector(lo, hi, 0, 1, 2, 3, 4, 5, 6, 7,
                                 8, 9, 10, 11, 12, 13, 14, 15);
}
__device__ __forceinline__ v16bf frag_b(const bf16* base, int lh) {
  // B 32x16 bf16 layout (stored [N][K]): VGPR0-3 = K 8h..8h+7,
  // VGPR4-7 = K 16+8h..16+8h+7  (h = lane>>4), same as A
  v8bf lo = *(const v8bf*)(base + 8 * lh);
  v8bf hi = *(const v8bf*)(base + 16 + 8 * lh);
  return __builtin_shufflevector(lo, hi, 0, 1, 2, 3, 4, 5, 6, 7,
                                 8, 9, 10, 11, 12, 13, 14, 15);
}

__global__ __launch_bounds__(256) void gemm_bf16_wmma_kernel(
    const bf16* __restrict__ A, const bf16* __restrict__ Bt,
    const float* __restrict__ bias, float* __restrict__ outF,
    bf16* __restrict__ outB, int M, int N, int K, float scale, int relu,
    long long aBatch, long long bBatch, long long cBatch) {
  __shared__ bf16 lA[BM * BK];
  __shared__ bf16 lB[BN * BK];

  const int tid  = threadIdx.x;
  const int lane = tid & 31;
  const int wave = tid >> 5;   // 0..7
  const int wm   = wave & 1;   // 2 waves in M
  const int wn   = wave >> 1;  // 4 waves in N
  const int lm   = lane & 15;
  const int lh   = lane >> 4;

  const long long z = blockIdx.z;
  const bf16* Ap = A + z * aBatch;
  const bf16* Bp = Bt + z * bBatch;

  const int m0 = blockIdx.y * BM;
  const int n0 = blockIdx.x * BN;

  // Global->LDS staging assignment (256 threads):
  //   A tile: 64 rows x 32 cols, 8 bf16 per thread
  //   B tile: 128 rows x 32 cols, 16 bf16 per thread
  const int arow = tid >> 2;
  const int acol = (tid & 3) * 8;
  const int brow = tid >> 1;
  const int bcol = (tid & 1) * 16;

  v8f acc[2][2];
#pragma unroll
  for (int i = 0; i < 2; ++i)
#pragma unroll
    for (int j = 0; j < 2; ++j)
#pragma unroll
      for (int r = 0; r < 8; ++r) acc[i][j][r] = 0.0f;

  const int nk = K / BK;
  const long long aGRow = (long long)(m0 + arow) * K;
  const long long bGRow = (long long)(n0 + brow) * K;

  // register-staged pipeline: preload k-tile 0
  v8bf ra  = *(const v8bf*)(Ap + aGRow + acol);
  v8bf rb0 = *(const v8bf*)(Bp + bGRow + bcol);
  v8bf rb1 = *(const v8bf*)(Bp + bGRow + bcol + 8);

  for (int kt = 0; kt < nk; ++kt) {
    *(v8bf*)(lA + arow * BK + acol)     = ra;
    *(v8bf*)(lB + brow * BK + bcol)     = rb0;
    *(v8bf*)(lB + brow * BK + bcol + 8) = rb1;
    __syncthreads();

    if (kt + 1 < nk) {
      const int k1 = (kt + 1) * BK;
      ra  = *(const v8bf*)(Ap + aGRow + k1 + acol);
      rb0 = *(const v8bf*)(Bp + bGRow + k1 + bcol);
      rb1 = *(const v8bf*)(Bp + bGRow + k1 + bcol + 8);
    }
    if (kt + 2 < nk) {  // global_prefetch_b8 for the tile after next
      const int k2 = (kt + 2) * BK;
      __builtin_prefetch(Ap + aGRow + k2 + acol, 0, 1);
      __builtin_prefetch(Bp + bGRow + k2 + bcol, 0, 1);
    }

    v16bf af[2], bfr[2];
#pragma unroll
    for (int i = 0; i < 2; ++i)
      af[i] = frag_a(lA + (wm * 32 + i * 16 + lm) * BK, lh);
#pragma unroll
    for (int j = 0; j < 2; ++j)
      bfr[j] = frag_b(lB + (wn * 32 + j * 16 + lm) * BK, lh);

#pragma unroll
    for (int i = 0; i < 2; ++i)
#pragma unroll
      for (int j = 0; j < 2; ++j)
        acc[i][j] = __builtin_amdgcn_wmma_f32_16x16x32_bf16(
            false, af[i], false, bfr[j], (short)0, acc[i][j], false, false);

    __syncthreads();
  }

  // epilogue: C layout — lane n = lane&15, VGPR r -> row r + 8*(lane>>4)
#pragma unroll
  for (int i = 0; i < 2; ++i) {
#pragma unroll
    for (int j = 0; j < 2; ++j) {
      const int gmb = m0 + wm * 32 + i * 16 + 8 * lh;
      const int gn  = n0 + wn * 32 + j * 16 + lm;
      const float bv = bias ? bias[gn] : 0.0f;
#pragma unroll
      for (int r = 0; r < 8; ++r) {
        float v = acc[i][j][r] * scale + bv;
        if (relu) v = fmaxf(v, 0.0f);
        const long long idx = z * cBatch + (long long)(gmb + r) * N + gn;
        if (outF) outF[idx] = v;
        if (outB) outB[idx] = (bf16)v;
      }
    }
  }
}

// ---------------------- row softmax (S=2048) -> bf16 ------------------------
__global__ __launch_bounds__(256) void softmax_bf16_kernel(
    const float* __restrict__ scores, bf16* __restrict__ attn, int S) {
  __shared__ float red[256];
  const long long row = blockIdx.x;
  const float* p = scores + row * (long long)S;
  bf16* o = attn + row * (long long)S;
  const int tid = threadIdx.x;
  float v[8];
  float mx = -1e30f;
#pragma unroll
  for (int j = 0; j < 8; ++j) {
    v[j] = p[tid + j * 256];
    mx = fmaxf(mx, v[j]);
  }
  red[tid] = mx;
  __syncthreads();
  for (int s = 128; s > 0; s >>= 1) {
    if (tid < s) red[tid] = fmaxf(red[tid], red[tid + s]);
    __syncthreads();
  }
  mx = red[0];
  __syncthreads();
  float sum = 0.0f;
#pragma unroll
  for (int j = 0; j < 8; ++j) {
    v[j] = __expf(v[j] - mx);
    sum += v[j];
  }
  red[tid] = sum;
  __syncthreads();
  for (int s = 128; s > 0; s >>= 1) {
    if (tid < s) red[tid] += red[tid + s];
    __syncthreads();
  }
  const float inv = 1.0f / red[0];
#pragma unroll
  for (int j = 0; j < 8; ++j) o[tid + j * 256] = (bf16)(v[j] * inv);
}

// ---------------- residual add + LayerNorm (E=1024 per row) -----------------
__global__ __launch_bounds__(256) void layernorm_residual_kernel(
    const float* __restrict__ xin, const float* __restrict__ yin,
    const float* __restrict__ gamma, const float* __restrict__ beta,
    float* __restrict__ outF, bf16* __restrict__ outB, int E) {
  __shared__ float redS[256];
  __shared__ float redQ[256];
  const long long row = blockIdx.x;
  const float* px = xin + row * (long long)E;
  const float* py = yin + row * (long long)E;
  const int tid = threadIdx.x;
  float v[4];
  float s = 0.0f, sq = 0.0f;
#pragma unroll
  for (int j = 0; j < 4; ++j) {
    const int i = tid + j * 256;
    v[j] = px[i] + py[i];
    s += v[j];
    sq += v[j] * v[j];
  }
  redS[tid] = s;
  redQ[tid] = sq;
  __syncthreads();
  for (int st = 128; st > 0; st >>= 1) {
    if (tid < st) {
      redS[tid] += redS[tid + st];
      redQ[tid] += redQ[tid + st];
    }
    __syncthreads();
  }
  const float mean = redS[0] / (float)E;
  const float var  = redQ[0] / (float)E - mean * mean;
  const float rstd = rsqrtf(var + 1e-5f);
#pragma unroll
  for (int j = 0; j < 4; ++j) {
    const int i = tid + j * 256;
    const float o = (v[j] - mean) * rstd * gamma[i] + beta[i];
    if (outF) outF[row * (long long)E + i] = o;
    if (outB) outB[row * (long long)E + i] = (bf16)o;
  }
}

// ---------------------------------------------------------------------------
extern "C" void kernel_launch(void* const* d_in, const int* in_sizes, int n_in,
                              void* d_out, int out_size, void* d_ws, size_t ws_size,
                              hipStream_t stream) {
  (void)in_sizes; (void)n_in; (void)out_size; (void)ws_size;
  const long long Bn = 8, S = 2048, E = 1024, F = 4096;
  const long long Mr = Bn * S;            // 16384 rows
  const long long ME = Mr * E;            // 16,777,216
  const long long SS = Bn * S * S;        // 33,554,432

  const float* x      = (const float*)d_in[0];
  const float* Wq     = (const float*)d_in[1];
  const float* bq     = (const float*)d_in[2];
  const float* Wk     = (const float*)d_in[3];
  const float* bk     = (const float*)d_in[4];
  const float* Wv     = (const float*)d_in[5];
  const float* bv     = (const float*)d_in[6];
  const float* gamma1 = (const float*)d_in[7];
  const float* beta1  = (const float*)d_in[8];
  const float* W1     = (const float*)d_in[9];
  const float* b1     = (const float*)d_in[10];
  const float* W2     = (const float*)d_in[11];
  const float* b2     = (const float*)d_in[12];
  const float* gamma2 = (const float*)d_in[13];
  const float* beta2  = (const float*)d_in[14];

  float* out_o = (float*)d_out;             // out
  float* out_k = out_o + ME;                // k
  float* out_q = out_o + 2 * ME;            // q
  float* out_v = out_o + 3 * ME;            // v

  size_t off = 0;
  auto carve = [&](size_t bytes) -> void* {
    void* p = (char*)d_ws + off;
    off += (bytes + 255) & ~(size_t)255;
    return p;
  };
  bf16*  xb      = (bf16*)carve(ME * 2);
  bf16*  WqT     = (bf16*)carve(E * E * 2);
  bf16*  WkT     = (bf16*)carve(E * E * 2);
  bf16*  WvT     = (bf16*)carve(E * E * 2);
  bf16*  W1T     = (bf16*)carve(E * F * 2);
  bf16*  W2T     = (bf16*)carve(E * F * 2);
  bf16*  qb      = (bf16*)carve(ME * 2);
  bf16*  kb      = (bf16*)carve(ME * 2);
  bf16*  vT      = (bf16*)carve(ME * 2);
  float* scores  = (float*)carve(SS * 4);
  bf16*  attnb   = (bf16*)carve(SS * 2);
  float* attn_o  = (float*)carve(ME * 4);
  float* h       = (float*)carve(ME * 4);
  bf16*  hb      = (bf16*)carve(ME * 2);
  bf16*  f1b     = (bf16*)carve(Mr * F * 2);
  float* ffn2    = (float*)carve(ME * 4);

  // 1) convert x, transpose+convert weights to bf16 [N][K]
  f32_to_bf16_kernel<<<dim3(ME / 1024), 256, 0, stream>>>(x, xb, ME);
  transpose_convert_kernel<<<dim3(E / 32, E / 32, 1), 256, 0, stream>>>(Wq, WqT, E, E, 0, 0);
  transpose_convert_kernel<<<dim3(E / 32, E / 32, 1), 256, 0, stream>>>(Wk, WkT, E, E, 0, 0);
  transpose_convert_kernel<<<dim3(E / 32, E / 32, 1), 256, 0, stream>>>(Wv, WvT, E, E, 0, 0);
  transpose_convert_kernel<<<dim3(F / 32, E / 32, 1), 256, 0, stream>>>(W1, W1T, E, F, 0, 0);
  transpose_convert_kernel<<<dim3(E / 32, F / 32, 1), 256, 0, stream>>>(W2, W2T, F, E, 0, 0);

  // 2) q/k/v projections (fp32 to outputs, bf16 copies for attention)
  dim3 gQKV(E / BN, Mr / BM, 1);
  gemm_bf16_wmma_kernel<<<gQKV, 256, 0, stream>>>(xb, WqT, bq, out_q, qb,
      (int)Mr, (int)E, (int)E, 1.0f, 0, 0, 0, 0);
  gemm_bf16_wmma_kernel<<<gQKV, 256, 0, stream>>>(xb, WkT, bk, out_k, kb,
      (int)Mr, (int)E, (int)E, 1.0f, 0, 0, 0, 0);
  gemm_bf16_wmma_kernel<<<gQKV, 256, 0, stream>>>(xb, WvT, bv, out_v, nullptr,
      (int)Mr, (int)E, (int)E, 1.0f, 0, 0, 0, 0);

  // 3) v^T per batch (fp32 v -> bf16 [E][S])
  transpose_convert_kernel<<<dim3(E / 32, S / 32, Bn), 256, 0, stream>>>(
      out_v, vT, (int)S, (int)E, S * E, E * S);

  // 4) scores = q @ k^T / sqrt(E)   (k rows are already the [N][K] layout)
  gemm_bf16_wmma_kernel<<<dim3(S / BN, S / BM, Bn), 256, 0, stream>>>(
      qb, kb, nullptr, scores, nullptr, (int)S, (int)S, (int)E,
      0.03125f, 0, S * E, S * E, S * S);

  // 5) softmax rows -> bf16 attention
  softmax_bf16_kernel<<<dim3(Bn * S), 256, 0, stream>>>(scores, attnb, (int)S);

  // 6) attn_out = attn @ v
  gemm_bf16_wmma_kernel<<<dim3(E / BN, S / BM, Bn), 256, 0, stream>>>(
      attnb, vT, nullptr, attn_o, nullptr, (int)S, (int)E, (int)S,
      1.0f, 0, S * S, E * S, S * E);

  // 7) h = LN(x + attn_out)
  layernorm_residual_kernel<<<dim3(Mr), 256, 0, stream>>>(
      x, attn_o, gamma1, beta1, h, hb, (int)E);

  // 8) ffn1 = relu(h @ W1 + b1) -> bf16 only
  gemm_bf16_wmma_kernel<<<dim3(F / BN, Mr / BM, 1), 256, 0, stream>>>(
      hb, W1T, b1, nullptr, f1b, (int)Mr, (int)F, (int)E, 1.0f, 1, 0, 0, 0);

  // 9) ffn2 = ffn1 @ W2 + b2
  gemm_bf16_wmma_kernel<<<dim3(E / BN, Mr / BM, 1), 256, 0, stream>>>(
      f1b, W2T, b2, ffn2, nullptr, (int)Mr, (int)E, (int)F, 1.0f, 0, 0, 0, 0);

  // 10) out = LN(h + ffn2)
  layernorm_residual_kernel<<<dim3(Mr), 256, 0, stream>>>(
      h, ffn2, gamma2, beta2, out_o, nullptr, (int)E);
}
